// HybridLinkPredictor_65180423684766
// MI455X (gfx1250) — compile-verified, matrix operand order checked by reading, not run on hardware
//
#include <hip/hip_runtime.h>
#include <cstdint>

typedef __attribute__((ext_vector_type(16))) _Float16 v16h;
typedef __attribute__((ext_vector_type(8)))  float    v8f;

// ---------------------------------------------------------------------------
// WMMA GEMM: out[M,C] = scale * (A[M,K] @ B[K,C] + bias1 + bias2), optional relu
//
// Block = 256 threads = 8 waves. All waves share one 64-column group; each
// wave owns one 16-row M tile (block covers 128 rows x 64 cols).
// B panel (K x 64) is cooperatively converted to f16 and staged in LDS in
// WMMA-fragment order: each lane's 16-half fragment is 32 contiguous bytes
// (ds_load_b128 x2). A is loaded as float4 pairs (global_load_b128) and
// converted with packed cvt. K-loop fully unrolled via template.
// ---------------------------------------------------------------------------
template <int K, int C>
__global__ __launch_bounds__(256) void gemm_wmma_f16(
    const float* __restrict__ A,
    const float* __restrict__ B,
    const float* __restrict__ bias1,
    const float* __restrict__ bias2,
    float* __restrict__ out,
    int M, float scale, int doRelu)
{
    constexpr int KSTEPS  = K / 32;
    constexpr int NGROUPS = C / 64;
    __shared__ _Float16 ldsB[K * 64];   // fragment-ordered B panel (<= 32 KB)

    const int nGroup  = blockIdx.x % NGROUPS;
    const int mBlock  = blockIdx.x / NGROUPS;
    const int colBase = nGroup * 64;
    const int tid     = threadIdx.x;
    const int mTiles  = (M + 15) >> 4;

    // ---- cooperative B fill: chunk = (kstep, nt, lane) -> 16 contiguous halves
    for (int chunk = tid; chunk < KSTEPS * 4 * 32; chunk += 256) {
        const int lane  = chunk & 31;
        const int nt    = (chunk >> 5) & 3;
        const int kstep = chunk >> 7;
        const int grp   = lane >> 4;
        const int col   = colBase + nt * 16 + (lane & 15);
        const int kb    = kstep * 32 + grp * 8;
        _Float16* dp = &ldsB[chunk * 16];
#pragma unroll
        for (int j = 0; j < 8; ++j) {
            dp[j]     = (_Float16)B[(kb + j) * C + col];
            dp[j + 8] = (_Float16)B[(kb + 16 + j) * C + col];
        }
    }
    __syncthreads();

    const int waveId = tid >> 5;
    const int lane   = tid & 31;
    const int mTile  = mBlock * 8 + waveId;
    if (mTile >= mTiles) return;                 // wave-uniform exit

    const int grp   = lane >> 4;                 // K-half select
    const int l15   = lane & 15;
    const int mBase = mTile * 16;

    int rowA = mBase + l15; if (rowA >= M) rowA = M - 1;   // ragged-tile clamp
    const float* __restrict__ Arow = A + (size_t)rowA * K;

    v8f acc[4] = {v8f{}, v8f{}, v8f{}, v8f{}};
    const v16h* __restrict__ fragB = (const v16h*)ldsB;

#pragma unroll
    for (int kstep = 0; kstep < KSTEPS; ++kstep) {
        const int ka = kstep * 32 + grp * 8;
        // ---- A fragment: 4x float4 (b128) loads, then packed converts
        const float4 p0 = *(const float4*)(Arow + ka);
        const float4 p1 = *(const float4*)(Arow + ka + 4);
        const float4 p2 = *(const float4*)(Arow + ka + 16);
        const float4 p3 = *(const float4*)(Arow + ka + 20);
        v16h a;
        a[0]  = (_Float16)p0.x; a[1]  = (_Float16)p0.y;
        a[2]  = (_Float16)p0.z; a[3]  = (_Float16)p0.w;
        a[4]  = (_Float16)p1.x; a[5]  = (_Float16)p1.y;
        a[6]  = (_Float16)p1.z; a[7]  = (_Float16)p1.w;
        a[8]  = (_Float16)p2.x; a[9]  = (_Float16)p2.y;
        a[10] = (_Float16)p2.z; a[11] = (_Float16)p2.w;
        a[12] = (_Float16)p3.x; a[13] = (_Float16)p3.y;
        a[14] = (_Float16)p3.z; a[15] = (_Float16)p3.w;
#pragma unroll
        for (int nt = 0; nt < 4; ++nt) {
            const v16h b = fragB[(kstep * 4 + nt) * 32 + lane];  // 32B contiguous
            acc[nt] = __builtin_amdgcn_wmma_f32_16x16x32_f16(
                false, a, false, b, (short)0, acc[nt], false, false);
        }
    }

    // ---- epilogue: C/D layout: col = lane&15, row = r + 8*(lane>>4)
#pragma unroll
    for (int nt = 0; nt < 4; ++nt) {
        const int col = colBase + nt * 16 + l15;
        float bv = 0.0f;
        if (bias1) bv += bias1[col];
        if (bias2) bv += bias2[col];
#pragma unroll
        for (int r = 0; r < 8; ++r) {
            const int row = mBase + r + grp * 8;
            if (row < M) {
                float v = acc[nt][r] + bv;
                if (doRelu) v = fmaxf(v, 0.0f);
                out[(size_t)row * C + col] = v * scale;
            }
        }
    }
}

// ---------------------------------------------------------------------------
// Degree / norm helpers
// ---------------------------------------------------------------------------
__global__ void fill_ones(float* __restrict__ p, int n)
{
    int i = blockIdx.x * blockDim.x + threadIdx.x;
    if (i < n) p[i] = 1.0f;                 // self-loop contributes 1 to degree
}

__global__ void degree_count(const int* __restrict__ dst, float* __restrict__ deg, int E)
{
    int e = blockIdx.x * blockDim.x + threadIdx.x;
    if (e < E) atomicAdd(&deg[dst[e]], 1.0f);
}

__global__ void rsqrt_deg(const float* __restrict__ deg, float* __restrict__ dinv, int n)
{
    int i = blockIdx.x * blockDim.x + threadIdx.x;
    if (i < n) {
        float d = deg[i];
        dinv[i] = (d > 0.0f) ? rsqrtf(d) : 0.0f;
    }
}

// out[i*C + c] = b[c]  (initialize accumulator with bias before scatter-add)
__global__ void bias_broadcast(float* __restrict__ out, const float* __restrict__ b,
                               int n, int C)
{
    int i = blockIdx.x * blockDim.x + threadIdx.x;
    if (i < n * C) out[i] = b[i % C];
}

__global__ void relu_inplace(float* __restrict__ x, int n)
{
    int i = blockIdx.x * blockDim.x + threadIdx.x;
    if (i < n) x[i] = fmaxf(x[i], 0.0f);
}

// ---------------------------------------------------------------------------
// GCN scatter-add: one wave per (edge or self-loop).
// Edge indices are wave-uniform -> scalar loads; lanes move float4/float2
// slices and atomically accumulate (global_atomic_add_f32, L2-resident).
// out[dst] += scale * dinv[src]*dinv[dst] * h[src]
// ---------------------------------------------------------------------------
__global__ void gcn_scatter(const float* __restrict__ h,
                            const int* __restrict__ src,
                            const int* __restrict__ dst,
                            const float* __restrict__ dinv,
                            float* __restrict__ out,
                            int E, int N, int C, float scale)
{
    const int wid  = (blockIdx.x * blockDim.x + threadIdx.x) >> 5;
    const int lane = threadIdx.x & 31;
    if (wid >= E + N) return;

    int s, d;
    if (wid < E) { s = src[wid]; d = dst[wid]; }
    else         { s = d = wid - E; }           // self-loop

    const float norm = dinv[s] * dinv[d] * scale;
    const float* __restrict__ hs = h   + (size_t)s * C;
    float* __restrict__       od = out + (size_t)d * C;

    if (C == 128) {
        const int c = lane * 4;
        float4 v = *(const float4*)(hs + c);
        atomicAdd(od + c + 0, norm * v.x);
        atomicAdd(od + c + 1, norm * v.y);
        atomicAdd(od + c + 2, norm * v.z);
        atomicAdd(od + c + 3, norm * v.w);
    } else {                                    // C == 64
        const int c = lane * 2;
        float2 v = *(const float2*)(hs + c);
        atomicAdd(od + c + 0, norm * v.x);
        atomicAdd(od + c + 1, norm * v.y);
    }
}

// ---------------------------------------------------------------------------
// Link-prediction decode: out[e] = dot(z[q0[e]], z[q1[e]]), OUT = 64
// ---------------------------------------------------------------------------
__global__ void edge_decode(const float* __restrict__ z,
                            const int* __restrict__ q0,
                            const int* __restrict__ q1,
                            float* __restrict__ out, int EQ)
{
    int e = blockIdx.x * blockDim.x + threadIdx.x;
    if (e >= EQ) return;
    const float4* __restrict__ za = (const float4*)(z + (size_t)q0[e] * 64);
    const float4* __restrict__ zb = (const float4*)(z + (size_t)q1[e] * 64);
    float acc = 0.0f;
#pragma unroll
    for (int i = 0; i < 16; ++i) {
        float4 a = za[i], b = zb[i];
        acc += a.x * b.x + a.y * b.y + a.z * b.z + a.w * b.w;
    }
    out[e] = acc;
}

// ---------------------------------------------------------------------------
static inline int cdiv(long long a, long long b) { return (int)((a + b - 1) / b); }

static void launch_gemm(const float* A, const float* B, const float* b1p,
                        const float* b2p, float* out, int M, int K, int C,
                        float scale, int doRelu, hipStream_t s)
{
    const int mTiles  = (M + 15) / 16;
    const int nGroups = C / 64;
    dim3 grid(cdiv(mTiles, 8) * nGroups), block(256);
    if (K == 128 && C == 128)
        gemm_wmma_f16<128, 128><<<grid, block, 0, s>>>(A, B, b1p, b2p, out, M, scale, doRelu);
    else if (K == 128 && C == 64)
        gemm_wmma_f16<128, 64><<<grid, block, 0, s>>>(A, B, b1p, b2p, out, M, scale, doRelu);
    else if (K == 128 && C == 256)
        gemm_wmma_f16<128, 256><<<grid, block, 0, s>>>(A, B, b1p, b2p, out, M, scale, doRelu);
    else if (K == 256 && C == 64)
        gemm_wmma_f16<256, 64><<<grid, block, 0, s>>>(A, B, b1p, b2p, out, M, scale, doRelu);
}

extern "C" void kernel_launch(void* const* d_in, const int* in_sizes, int n_in,
                              void* d_out, int out_size, void* d_ws, size_t ws_size,
                              hipStream_t stream)
{
    const int IN = 128, HID = 128, OUT = 64, FCH = 256;
    const float ALPHA = 0.5f;

    const float* x   = (const float*)d_in[0];
    const int*   ei  = (const int*)  d_in[1];   // [2, E]
    const int*   eli = (const int*)  d_in[2];   // [2, EQ]
    const float* W1  = (const float*)d_in[3];
    const float* b1  = (const float*)d_in[4];
    const float* W2  = (const float*)d_in[5];
    const float* b2  = (const float*)d_in[6];
    const float* Wf1 = (const float*)d_in[7];
    const float* bf1 = (const float*)d_in[8];
    const float* Wf2 = (const float*)d_in[9];
    const float* bf2 = (const float*)d_in[10];
    float* outp = (float*)d_out;

    const int N  = in_sizes[0] / IN;
    const int E  = in_sizes[1] / 2;
    const int EQ = in_sizes[2] / 2;

    const int* src = ei;            // edge_index[0]
    const int* dst = ei + E;        // edge_index[1]
    const int* q0  = eli;
    const int* q1  = eli + EQ;

    // ---- workspace carve (f32) ----
    float* ws   = (float*)d_ws;
    float* deg  = ws;                           size_t off = (size_t)N;
    float* dinv = ws + off;                     off += (size_t)N;
    float* h1   = ws + off;                     off += (size_t)N * HID;  // x@W1
    float* a1   = ws + off;                     off += (size_t)N * HID;  // agg1+relu
    float* h2   = ws + off;                     off += (size_t)N * OUT;  // a1@W2
    float* f1   = ws + off;                     off += (size_t)N * FCH;  // fc hidden
    float* z    = ws + off;                     off += (size_t)N * OUT;  // final embed
    (void)ws_size;

    const int TB = 256;

    // 1) degree with self-loops, 2) dinv = rsqrt(deg)
    fill_ones<<<cdiv(N, TB), TB, 0, stream>>>(deg, N);
    degree_count<<<cdiv(E, TB), TB, 0, stream>>>(dst, deg, E);
    rsqrt_deg<<<cdiv(N, TB), TB, 0, stream>>>(deg, dinv, N);

    // 3) h1 = x @ W1 (pre-aggregation, no bias)
    launch_gemm(x, W1, nullptr, nullptr, h1, N, IN, HID, 1.0f, 0, stream);

    // 4) a1 = b1; a1[dst] += norm * h1[src]; relu
    bias_broadcast<<<cdiv((long long)N * HID, TB), TB, 0, stream>>>(a1, b1, N, HID);
    gcn_scatter<<<cdiv((long long)(E + N) * 32, TB), TB, 0, stream>>>(
        h1, src, dst, dinv, a1, E, N, HID, 1.0f);
    relu_inplace<<<cdiv((long long)N * HID, TB), TB, 0, stream>>>(a1, N * HID);

    // 5) h2 = a1 @ W2 (pre-aggregation, no bias)
    launch_gemm(a1, W2, nullptr, nullptr, h2, N, HID, OUT, 1.0f, 0, stream);

    // 6) f1 = relu(x @ Wf1 + bf1)
    launch_gemm(x, Wf1, bf1, nullptr, f1, N, IN, FCH, 1.0f, 1, stream);

    // 7) z = ALPHA*(f1@Wf2 + bf2) + (1-ALPHA)*b2   [GCN bias folded in; ALPHA=0.5]
    launch_gemm(f1, Wf2, bf2, b2, z, N, FCH, OUT, ALPHA, 0, stream);

    // 8) z += (1-ALPHA) * norm * h2[src]  (scatter with 0.5 folded into scale)
    gcn_scatter<<<cdiv((long long)(E + N) * 32, TB), TB, 0, stream>>>(
        h2, src, dst, dinv, z, E, N, OUT, 1.0f - ALPHA);

    // 9) decode query edges
    edge_decode<<<cdiv(EQ, TB), TB, 0, stream>>>(z, q0, q1, outp, EQ);
}
